// Self_Attn_15702400434266
// MI455X (gfx1250) — matrix operand order repaired and checked by reading, not compiled, not run on hardware
//
#include <hip/hip_runtime.h>
#include <hip/hip_bf16.h>

// ---------------------------------------------------------------------------
// Self-attention (SAGAN-style) for MI455X / gfx1250, wave32 + WMMA f16.
// B=16, C=512, CT=64, N=48*48=2304.
// ---------------------------------------------------------------------------

typedef __attribute__((ext_vector_type(16))) _Float16 v16h;
typedef __attribute__((ext_vector_type(8)))  float    v8f;
typedef __attribute__((ext_vector_type(4)))  int      v4i;

#define BB   16
#define CC   512
#define CTD  64
#define NN   2304      // H*W
#define NB128 18       // NN / 128
#define NB64  36       // NN / 64

#if __has_builtin(__builtin_amdgcn_global_load_async_to_lds_b128)
#define HAVE_ASYNC_LDS 1
typedef __attribute__((address_space(1))) v4i* as1_v4i_p;
typedef __attribute__((address_space(3))) v4i* as3_v4i_p;
#else
#define HAVE_ASYNC_LDS 0
#endif

// ---- WMMA wrapper ---------------------------------------------------------
__device__ __forceinline__ v8f wmma16(v16h a, v16h b, v8f c) {
  // D = A(16x32 f16) x B(32x16 f16) + C(16x16 f32)
  return __builtin_amdgcn_wmma_f32_16x16x32_f16(
      /*neg_a=*/false, a, /*neg_b=*/false, b,
      /*c_mod=*/(short)0, c, /*reuse_a=*/false, /*reuse_b=*/false);
}

// ---- A fragment (16x32 f16, row-major source, ld multiple of 8 halfs) -----
// ISA 7.12.2: lane L holds row M = L%16; halfs 0..7 -> K = b1+0..7,
// halfs 8..15 -> K = 16+b1+0..7 with b1 = (L<16)?0:8. Two 16B loads.
__device__ __forceinline__ v16h load_a16(const _Float16* base, int ld, int lane) {
  const int M  = lane & 15;
  const int b1 = (lane & 16) ? 8 : 0;
  const _Float16* p = base + M * ld + b1;
  union { v16h v; uint4 u[2]; } t;
  t.u[0] = *(const uint4*)(p);
  t.u[1] = *(const uint4*)(p + 16);
  return t.v;
}

// ---- B fragment (32x16 f16) from K-contiguous (transposed) storage --------
// Source T[n][k] row-major (ld halfs, mult. of 8). Lane L: column N = L%16,
// halfs 0..7 -> K = kb+0..7, halfs 8..15 -> K = 16+kb+0..7, kb = (L<16)?0:8.
__device__ __forceinline__ v16h load_b16(const _Float16* base, int ld, int lane) {
  const int Nn = lane & 15;
  const int kb = (lane & 16) ? 8 : 0;
  const _Float16* p = base + Nn * ld + kb;
  union { v16h v; uint4 u[2]; } t;
  t.u[0] = *(const uint4*)(p);
  t.u[1] = *(const uint4*)(p + 16);
  return t.v;
}

__device__ __forceinline__ v8f vzero8() {
  v8f z = {0.f, 0.f, 0.f, 0.f, 0.f, 0.f, 0.f, 0.f};
  return z;
}

__device__ __forceinline__ void wait_async_lds() {
#if HAVE_ASYNC_LDS
#if __has_builtin(__builtin_amdgcn_s_wait_asynccnt)
  __builtin_amdgcn_s_wait_asynccnt(0);
#else
  asm volatile("s_wait_asynccnt 0" ::: "memory");
#endif
#endif
}

// ===========================================================================
// Kernel 0: convert the four tiny weight matrices to f16 (L2 resident).
// Wq/Wk/Wv: (64,512), Wo: (512,64) -> 4 * 32768 elements.
// ===========================================================================
__global__ __launch_bounds__(256) void cvt_w_kernel(
    const float* __restrict__ wq, const float* __restrict__ wk,
    const float* __restrict__ wv, const float* __restrict__ wo,
    _Float16* __restrict__ whq, _Float16* __restrict__ whk,
    _Float16* __restrict__ whv, _Float16* __restrict__ who) {
  int i = blockIdx.x * 256 + threadIdx.x;   // grid = 512 blocks
  int seg = i >> 15;
  int off = i & 32767;
  const float* s = (seg == 0) ? wq : (seg == 1) ? wk : (seg == 2) ? wv : wo;
  _Float16*    d = (seg == 0) ? whq : (seg == 1) ? whk : (seg == 2) ? whv : who;
  d[off] = (_Float16)s[off];
}

// ===========================================================================
// Kernel 1: QKV projection.  q/k/v[b,ct,n] = W{q,k,v}[ct,:] . x[b,:,n] + bias
// Block tile: CT(64) x 128 n-columns, all three outputs at once.
// 8 waves: wave w -> row subtile rt=w%4, col subtiles (w/4)*4 + {0..3}.
// x tile (32 c x 128 n) staged f32->f16 *transposed* in LDS so B fragments
// are K-contiguous.  16 k-steps over C=512; next tile prefetched.
// ===========================================================================
__global__ __launch_bounds__(256) void qkv_kernel(
    const float* __restrict__ x,          // (B, C, N) f32
    const _Float16* __restrict__ whq,     // (CT, C) f16
    const _Float16* __restrict__ whk,
    const _Float16* __restrict__ whv,
    const float* __restrict__ bq, const float* __restrict__ bk,
    const float* __restrict__ bv,
    _Float16* __restrict__ qh,            // (B, CT, N) f16
    _Float16* __restrict__ kh,
    _Float16* __restrict__ vh) {
  const int b   = blockIdx.x / NB128;
  const int n0  = (blockIdx.x % NB128) * 128;
  const int tid  = threadIdx.x;
  const int wave = tid >> 5;
  const int lane = tid & 31;
  const int rt   = wave & 3;   // ct-row subtile
  const int chh  = wave >> 2;  // column half (4 subtiles each)

  __shared__ __align__(16) _Float16 xt[128][40];  // [n][c], ld=40 (pad, 16B ok)

  v8f accq[4], acck[4], accv[4];
  for (int j = 0; j < 4; ++j) { accq[j] = vzero8(); acck[j] = vzero8(); accv[j] = vzero8(); }

  const float* xb = x + (size_t)b * CC * NN;

  for (int c0 = 0; c0 < CC; c0 += 32) {
    __syncthreads();
    // prefetch next k-step's x tile into caches (global_prefetch_b8)
    if (c0 + 32 < CC) {
      const float* nx = xb + (size_t)(c0 + 32 + (tid >> 3)) * NN + n0 + (tid & 7) * 16;
      __builtin_prefetch(nx, 0, 1);
    }
    for (int i = tid; i < 32 * 128; i += 256) {
      int cc = i >> 7;     // 0..31
      int nn = i & 127;    // coalesced along nn
      xt[nn][cc] = (_Float16)xb[(size_t)(c0 + cc) * NN + n0 + nn];
    }
    __syncthreads();

    v16h aq = load_a16(whq + (size_t)(rt * 16) * CC + c0, CC, lane);
    v16h ak = load_a16(whk + (size_t)(rt * 16) * CC + c0, CC, lane);
    v16h av = load_a16(whv + (size_t)(rt * 16) * CC + c0, CC, lane);
    for (int j = 0; j < 4; ++j) {
      v16h bf = load_b16(&xt[(chh * 4 + j) * 16][0], 40, lane);
      accq[j] = wmma16(aq, bf, accq[j]);
      acck[j] = wmma16(ak, bf, acck[j]);
      accv[j] = wmma16(av, bf, accv[j]);
    }
  }

  // epilogue: bias + f16 store (D layout: reg r, lane L -> M = r + 8*(L/16))
  const int ml = lane & 15;
  const int rh = (lane >> 4) << 3;
  for (int j = 0; j < 4; ++j) {
    int n = n0 + (chh * 4 + j) * 16 + ml;
    for (int r = 0; r < 8; ++r) {
      int ct = rt * 16 + rh + r;
      size_t off = ((size_t)b * CTD + ct) * NN + n;
      qh[off] = (_Float16)(accq[j][r] + bq[ct]);
      kh[off] = (_Float16)(acck[j][r] + bk[ct]);
      vh[off] = (_Float16)(accv[j][r] + bv[ct]);
    }
  }
}

// ===========================================================================
// Kernel 2: flash attention over n (softmax axis), per (batch, 64-wide m-blk).
//   S[n,m] = sum_c q[c,n] k[c,m]    (K = CT = 64, two WMMA k-steps)
//   beta   = softmax_n(S)           (online, parallel across all 256 thr)
//   O[c,m] = sum_n v[c,n] beta[n,m] (accumulated in registers, rescaled)
// V tile staged via GLOBAL_LOAD_ASYNC_TO_LDS (ASYNCcnt) when available.
// ===========================================================================
__global__ __launch_bounds__(256) void attn_kernel(
    const _Float16* __restrict__ qh,   // (B, CT, N)
    const _Float16* __restrict__ kh,
    const _Float16* __restrict__ vh,
    _Float16* __restrict__ oh) {       // (B, CT, N)
  const int b   = blockIdx.x / NB64;
  const int m0  = (blockIdx.x % NB64) * 64;
  const int tid  = threadIdx.x;
  const int wave = tid >> 5;
  const int lane = tid & 31;
  const int rt   = wave & 3;   // n-subtile (phase S) / c-subtile (phase O)
  const int chh  = wave >> 2;  // owns m-subtiles chh*2 + {0,1}

  __shared__ __align__(16) _Float16 Kt[64][72];  // [m][c] transposed
  __shared__ __align__(16) _Float16 Qt[64][72];  // [n][c] transposed
  __shared__ __align__(16) _Float16 Vr[64][72];  // [c][n] row-major
  __shared__ __align__(16) _Float16 Pt[64][72];  // [m][n] transposed exp(S)
  __shared__ float Sb[64][68];                   // [n][m] f32 scores
  __shared__ float pmax[4][64];
  __shared__ float psum[4][64];
  __shared__ float nmax_s[64];
  __shared__ float scale_s[64];
  __shared__ float rsum_s[64];

  // stage K block once (coalesced along m)
  for (int i = tid; i < 64 * 64; i += 256) {
    int cc = i >> 6, mm = i & 63;
    Kt[mm][cc] = kh[((size_t)b * CTD + cc) * NN + m0 + mm];
  }

  v8f acco[2] = {vzero8(), vzero8()};
  float run_max = -1e30f, run_sum = 0.f;   // live in threads tid<64 (col m=tid)

  const int sm   = tid & 63;   // softmax column owned in parallel phases
  const int part = tid >> 6;   // n-range part*16 .. part*16+15

  for (int n0 = 0; n0 < NN; n0 += 64) {
    __syncthreads();   // prior O-phase reads of Vr/Pt done; Kt visible (1st it)

#if HAVE_ASYNC_LDS
    // V tile is a straight copy: use the async global->LDS path (ASYNCcnt).
    for (int i = tid; i < 512; i += 256) {   // 64 rows x 8 x 16B
      int cc = i >> 3, chk = i & 7;
      const _Float16* g = vh + ((size_t)b * CTD + cc) * NN + n0 + chk * 8;
      __builtin_amdgcn_global_load_async_to_lds_b128(
          (as1_v4i_p)(uintptr_t)g,
          (as3_v4i_p)(uintptr_t)(&Vr[cc][chk * 8]),
          0, 0);
    }
    for (int i = tid; i < 64 * 64; i += 256) {
      int cc = i >> 6, nn = i & 63;
      Qt[nn][cc] = qh[((size_t)b * CTD + cc) * NN + n0 + nn];
    }
    wait_async_lds();
#else
    for (int i = tid; i < 64 * 64; i += 256) {
      int cc = i >> 6, nn = i & 63;
      Qt[nn][cc] = qh[((size_t)b * CTD + cc) * NN + n0 + nn];
      Vr[cc][nn] = vh[((size_t)b * CTD + cc) * NN + n0 + nn];
    }
#endif
    __syncthreads();

    // ---- S = Q^T K : 64(n) x 64(m); A fragment hoisted over m-subtiles ----
    {
      v8f s0 = vzero8(), s1 = vzero8();
      for (int k0 = 0; k0 < 64; k0 += 32) {
        v16h a  = load_a16(&Qt[rt * 16][k0], 72, lane);          // A[n][c]
        v16h b0 = load_b16(&Kt[(chh * 2) * 16][k0], 72, lane);   // B[c][m]
        v16h b1 = load_b16(&Kt[(chh * 2 + 1) * 16][k0], 72, lane);
        s0 = wmma16(a, b0, s0);
        s1 = wmma16(a, b1, s1);
      }
      int ml = lane & 15, rh = (lane >> 4) << 3;
      for (int r = 0; r < 8; ++r) {
        Sb[rt * 16 + rh + r][(chh * 2) * 16 + ml]     = s0[r];
        Sb[rt * 16 + rh + r][(chh * 2 + 1) * 16 + ml] = s1[r];
      }
    }
    __syncthreads();

    // ---- softmax over n: partial max across 4 thread groups ----
    {
      float t0 = -1e30f;
      for (int i = 0; i < 16; ++i) t0 = fmaxf(t0, Sb[part * 16 + i][sm]);
      pmax[part][sm] = t0;
    }
    __syncthreads();
    if (tid < 64) {
      float tmax = fmaxf(fmaxf(pmax[0][tid], pmax[1][tid]),
                         fmaxf(pmax[2][tid], pmax[3][tid]));
      float nmax = fmaxf(run_max, tmax);
      scale_s[tid] = __expf(run_max - nmax);
      nmax_s[tid] = nmax;
      run_max = nmax;
    }
    __syncthreads();
    // ---- exp + packed Pt write + partial sums (all 256 threads) ----
    {
      float nmax = nmax_s[sm];
      float tsum = 0.f;
      union { uint4 u; _Float16 h[8]; } pk0, pk1;
      for (int i = 0; i < 8; ++i) {
        float p = __expf(Sb[part * 16 + i][sm] - nmax);
        pk0.h[i] = (_Float16)p; tsum += p;
      }
      for (int i = 0; i < 8; ++i) {
        float p = __expf(Sb[part * 16 + 8 + i][sm] - nmax);
        pk1.h[i] = (_Float16)p; tsum += p;
      }
      *(uint4*)&Pt[sm][part * 16]     = pk0.u;
      *(uint4*)&Pt[sm][part * 16 + 8] = pk1.u;
      psum[part][sm] = tsum;
    }
    __syncthreads();
    if (tid < 64) {
      run_sum = run_sum * scale_s[tid] +
                (psum[0][tid] + psum[1][tid]) + (psum[2][tid] + psum[3][tid]);
    }

    // ---- O = O*scale + V @ P : 64(c) x 64(m); A hoisted over m-subtiles ----
    {
      int ml = lane & 15;
      float sc0 = scale_s[(chh * 2) * 16 + ml];
      float sc1 = scale_s[(chh * 2 + 1) * 16 + ml];
      v8f a0 = acco[0], a1 = acco[1];
      for (int r = 0; r < 8; ++r) { a0[r] *= sc0; a1[r] *= sc1; }
      for (int k0 = 0; k0 < 64; k0 += 32) {
        v16h a  = load_a16(&Vr[rt * 16][k0], 72, lane);          // A[c][n]
        v16h b0 = load_b16(&Pt[(chh * 2) * 16][k0], 72, lane);   // B[n][m]
        v16h b1 = load_b16(&Pt[(chh * 2 + 1) * 16][k0], 72, lane);
        a0 = wmma16(a, b0, a0);
        a1 = wmma16(a, b1, a1);
      }
      acco[0] = a0; acco[1] = a1;
    }
  }

  if (tid < 64) rsum_s[tid] = run_sum;
  __syncthreads();

  {
    int ml = lane & 15, rh = (lane >> 4) << 3;
    for (int j = 0; j < 2; ++j) {
      int mt = chh * 2 + j;
      int m = m0 + mt * 16 + ml;
      float inv = 1.f / rsum_s[mt * 16 + ml];
      for (int r = 0; r < 8; ++r) {
        int cc = rt * 16 + rh + r;
        oh[((size_t)b * CTD + cc) * NN + m] = (_Float16)(acco[j][r] * inv);
      }
    }
  }
}

// ===========================================================================
// Kernel 3: output projection + residual.
//   y[b,c,n] = gamma * (Wo[c,:] . o[:,n] + bo[c]) + x[b,c,n]
// Block tile 64(c) x 128(n); K = CT = 64 (two WMMA k-steps).
// ===========================================================================
__global__ __launch_bounds__(256) void out_kernel(
    const _Float16* __restrict__ who,  // (C, CT) f16
    const _Float16* __restrict__ oh,   // (B, CT, N) f16
    const float* __restrict__ bo,
    const float* __restrict__ gamma,
    const float* __restrict__ x,       // (B, C, N) f32
    float* __restrict__ y) {
  int t = blockIdx.x;
  const int nblk = t % NB128; t /= NB128;
  const int cblk = t % 8;
  const int b    = t / 8;
  const int n0 = nblk * 128;
  const int c0 = cblk * 64;
  const int tid  = threadIdx.x;
  const int wave = tid >> 5;
  const int lane = tid & 31;
  const int rt   = wave & 3;
  const int chh  = wave >> 2;

  __shared__ __align__(16) _Float16 Ot[128][72];  // [n][ct] transposed

  for (int i = tid; i < 64 * 128; i += 256) {
    int cc = i >> 7, nn = i & 127;
    Ot[nn][cc] = oh[((size_t)b * CTD + cc) * NN + n0 + nn];
  }
  __syncthreads();

  v8f acc[4] = {vzero8(), vzero8(), vzero8(), vzero8()};
  for (int k0 = 0; k0 < 64; k0 += 32) {
    v16h a = load_a16(who + (size_t)(c0 + rt * 16) * CTD + k0, CTD, lane);
    for (int j = 0; j < 4; ++j) {
      v16h bb = load_b16(&Ot[(chh * 4 + j) * 16][k0], 72, lane);
      acc[j] = wmma16(a, bb, acc[j]);
    }
  }

  const float g = gamma[0];
  const int ml = lane & 15, rh = (lane >> 4) << 3;
  for (int j = 0; j < 4; ++j) {
    int n = n0 + (chh * 4 + j) * 16 + ml;
    for (int r = 0; r < 8; ++r) {
      int cc = c0 + rt * 16 + rh + r;
      size_t off = ((size_t)b * CC + cc) * NN + n;
      y[off] = g * (acc[j][r] + bo[cc]) + x[off];
    }
  }
}

// ===========================================================================
extern "C" void kernel_launch(void* const* d_in, const int* in_sizes, int n_in,
                              void* d_out, int out_size, void* d_ws, size_t ws_size,
                              hipStream_t stream) {
  (void)in_sizes; (void)n_in; (void)out_size; (void)ws_size;

  const float* x     = (const float*)d_in[0];
  const float* Wq    = (const float*)d_in[1];
  const float* bq    = (const float*)d_in[2];
  const float* Wk    = (const float*)d_in[3];
  const float* bk    = (const float*)d_in[4];
  const float* Wv    = (const float*)d_in[5];
  const float* bv    = (const float*)d_in[6];
  const float* Wo    = (const float*)d_in[7];
  const float* bo    = (const float*)d_in[8];
  const float* gamma = (const float*)d_in[9];
  float* y = (float*)d_out;

  // workspace layout (f16): weights + q,k,v,o  (~19.1 MB)
  const size_t WSZ = 32768;                 // 64*512
  const size_t NQ  = (size_t)BB * CTD * NN; // 2,359,296
  _Float16* whq = (_Float16*)d_ws;
  _Float16* whk = whq + WSZ;
  _Float16* whv = whk + WSZ;
  _Float16* who = whv + WSZ;
  _Float16* qh  = who + WSZ;
  _Float16* kh  = qh + NQ;
  _Float16* vh  = kh + NQ;
  _Float16* oh  = vh + NQ;

  cvt_w_kernel<<<512, 256, 0, stream>>>(Wq, Wk, Wv, Wo, whq, whk, whv, who);
  qkv_kernel<<<BB * NB128, 256, 0, stream>>>(x, whq, whk, whv, bq, bk, bv,
                                             qh, kh, vh);
  attn_kernel<<<BB * NB64, 256, 0, stream>>>(qh, kh, vh, oh);
  out_kernel<<<BB * 8 * NB128, 256, 0, stream>>>(who, oh, bo, gamma, x, y);
}
